// SepConvGRU_54597624266888
// MI455X (gfx1250) — matrix operand order repaired and checked by reading, not compiled, not run on hardware
//
#include <hip/hip_runtime.h>
#include <stdint.h>
#include <stddef.h>

#define BB   16
#define CH   128
#define CX   320
#define CIN  448
#define HH   64
#define WW   160
#define HP   (HH+4)           // padded height
#define WP   (WW+4)           // padded width
#define HW   (HH*WW)          // 10240
#define NSP  (BB*HW)          // 163840
#define KTOT (CIN*5)          // 2240
#define NKB  (KTOT/32)        // 70
#define KBPT (CIN/32)         // 14 k-blocks per tap
#define NTSTRIDE 528          // padded LDS n-tile stride (u16 units)

typedef __attribute__((ext_vector_type(16))) __bf16    v16bf;
typedef __attribute__((ext_vector_type(8)))  float     v8f;
typedef __attribute__((ext_vector_type(4)))  float     f32x4;
typedef __attribute__((ext_vector_type(8)))  uint16_t  u16x8;

__device__ __forceinline__ uint16_t f2bf(float f) {
  uint32_t u = __float_as_uint(f);
  u += 0x7FFFu + ((u >> 16) & 1u);
  return (uint16_t)(u >> 16);
}
__device__ __forceinline__ float sigmoidf_(float v) {
  return 1.0f / (1.0f + __expf(-v));
}

// one per-lane 16B async copy global -> LDS (ASYNCcnt-tracked)
__device__ __forceinline__ void async_cp16(const uint16_t* g, uint32_t ldsByteOff) {
  asm volatile("global_load_async_to_lds_b128 %0, %1, off"
               :: "v"(ldsByteOff), "v"((uint64_t)(uintptr_t)g)
               : "memory");
}
__device__ __forceinline__ void wait_async_le(int n) {
  if (n == 0)      asm volatile("s_wait_asynccnt 0x0" ::: "memory");
  else if (n == 1) asm volatile("s_wait_asynccnt 0x1" ::: "memory");
  else             asm volatile("s_wait_asynccnt 0x2" ::: "memory");
}

// B-fragment LDS index (u16 units) for element group (kgrp*8 .. +7) of column nl
__device__ __forceinline__ int bfrag_idx(int nl, int kgrp) {
  return (nl >> 4) * NTSTRIDE + (16 * (kgrp & 1) + (nl & 15)) * 16 + (kgrp >> 1) * 8;
}

// ---------------------------------------------------------------------------
// Weight repack: fp32 [O,448,5] -> per-lane A-fragment bf16 buffer
// aFrag[(((kb*MT)+mt)*32+lane)*16+e], CDNA5 16-bit A 16x32 lane layout.
// ---------------------------------------------------------------------------
__global__ void pack_wfrag(const float* __restrict__ w0, const float* __restrict__ w1,
                           uint16_t* __restrict__ aFrag, int MT) {
  int gid = blockIdx.x * 256 + threadIdx.x;
  int total = NKB * MT * 32;
  if (gid >= total) return;
  int lane = gid & 31;
  int mt   = (gid >> 5) % MT;
  int kb   = gid / (32 * MT);
  int m    = mt * 16 + (lane & 15);
  const float* wsrc = (m < CH) ? w0 : w1;
  int mm = m & (CH - 1);
  uint16_t* dst = aFrag + (size_t)gid * 16;
#pragma unroll
  for (int e = 0; e < 16; ++e) {
    int v = e >> 1, p = e & 1;
    int klocal = ((lane < 16) ? 0 : 8) + ((v & 3) << 1) + p + ((v >= 4) ? 16 : 0);
    int K = kb * 32 + klocal;
    int t = K / CIN;
    int c = K % CIN;
    dst[e] = f2bf(wsrc[((size_t)mm * CIN + c) * 5 + t]);
  }
}

// NCHW f32 (h | x) -> zero-padded NHWC bf16 act buffer
__global__ void pack_act(const float* __restrict__ h, const float* __restrict__ x,
                         uint16_t* __restrict__ act) {
  int n  = blockIdx.x * 256 + threadIdx.x;     // spatial
  int cc = blockIdx.y * 8;                     // channel chunk
  int b = n / HW, rem = n - b * HW;
  int y = rem / WW, xx = rem - y * WW;
  size_t ps = ((size_t)(b * HP + y + 2)) * WP + (xx + 2);
  u16x8 v;
#pragma unroll
  for (int r = 0; r < 8; ++r) {
    int c = cc + r;
    float f = (c < CH) ? h[((size_t)b * CH + c) * HW + rem]
                       : x[((size_t)b * CX + (c - CH)) * HW + rem];
    v[r] = f2bf(f);
  }
  *(u16x8*)&act[ps * CIN + cc] = v;
}

// NCHW f32 h -> NHWC f32 hF
__global__ void pack_hF(const float* __restrict__ h, float* __restrict__ hF) {
  int n  = blockIdx.x * 256 + threadIdx.x;
  int cc = blockIdx.y * 8;
  int b = n / HW, rem = n - b * HW;
  f32x4 a, bv;
#pragma unroll
  for (int r = 0; r < 4; ++r) a[r]  = h[((size_t)b * CH + cc + r) * HW + rem];
#pragma unroll
  for (int r = 0; r < 4; ++r) bv[r] = h[((size_t)b * CH + cc + 4 + r) * HW + rem];
  float* dp = hF + (size_t)n * CH + cc;
  *(f32x4*)dp = a;
  *(f32x4*)(dp + 4) = bv;
}

// ---------------------------------------------------------------------------
// Fused z/r conv GEMM: M=256 (z||r), N=64, K=2240. Async double-buffered LDS.
// ---------------------------------------------------------------------------
template <int VERT>
__global__ __launch_bounds__(256)
void gemm_zr(const uint16_t* __restrict__ act, const uint16_t* __restrict__ aFrag,
             const float* __restrict__ bz, const float* __restrict__ br,
             const float* __restrict__ hF,
             float* __restrict__ zbuf, uint16_t* __restrict__ rh) {
  __shared__ alignas(32) uint16_t lds[2][4 * NTSTRIDE];
  const int tid  = threadIdx.x;
  const int lane = tid & 31;
  const int wave = tid >> 5;
  const int n0   = blockIdx.x * 64;

  // staging chunk: (kgrp 0..3, nl 0..63) -> one b128 per thread per k-block
  const int kgrp = tid >> 6;
  const int nl   = tid & 63;
  {
    // precompute thread global base + both LDS destinations
  }
  const int n  = n0 + nl;
  const int b  = n / HW;
  const int rm = n - b * HW;
  const int y  = rm / WW;
  const int xx = rm - y * WW;
  const size_t ps = ((size_t)(b * HP + y + 2)) * WP + (xx + 2);
  const uint16_t* gbase = act + ps * CIN + kgrp * 8;
  const int fidx = bfrag_idx(nl, kgrp);
  uint32_t lb0 = (uint32_t)(uintptr_t)&lds[0][fidx];
  uint32_t lb1 = (uint32_t)(uintptr_t)&lds[1][fidx];

  v8f acc[2][4];
#pragma unroll
  for (int i = 0; i < 2; ++i)
#pragma unroll
    for (int j = 0; j < 4; ++j) acc[i][j] = (v8f){0,0,0,0,0,0,0,0};

  const v16bf* ag = reinterpret_cast<const v16bf*>(aFrag);

  auto koff = [&](int kb) -> int {
    int t  = kb / KBPT;
    int c0 = (kb - t * KBPT) * 32;
    int sh = (t - 2) * (VERT ? WP : 1);
    return c0 + sh * CIN;                    // elements
  };

  async_cp16(gbase + koff(0), lb0);          // prologue stage kb=0 -> buf0

  for (int kb = 0; kb < NKB; ++kb) {
    const int cur = kb & 1;
    if (kb + 1 < NKB) {
      async_cp16(gbase + koff(kb + 1), cur ? lb0 : lb1);
      wait_async_le(1);                      // previous batch landed
    } else {
      wait_async_le(0);
    }
    __syncthreads();

    const uint16_t* lcur = lds[cur];
    v16bf a0 = ag[((size_t)kb * 16 + wave * 2 + 0) * 32 + lane];
    v16bf a1 = ag[((size_t)kb * 16 + wave * 2 + 1) * 32 + lane];
#pragma unroll
    for (int nt = 0; nt < 4; ++nt) {
      v16bf bf = *reinterpret_cast<const v16bf*>(&lcur[nt * NTSTRIDE + lane * 16]);
      acc[0][nt] = __builtin_amdgcn_wmma_f32_16x16x32_bf16(
          false, a0, false, bf, (short)0, acc[0][nt], false, false);
      acc[1][nt] = __builtin_amdgcn_wmma_f32_16x16x32_bf16(
          false, a1, false, bf, (short)0, acc[1][nt], false, false);
    }
  }

  // epilogue: z -> zbuf (NHWC f32); r*h -> rh (padded NHWC bf16)
#pragma unroll
  for (int i = 0; i < 2; ++i) {
    const int mt = wave * 2 + i;                       // wave-uniform
    const int m0 = mt * 16 + ((lane >= 16) ? 8 : 0);
#pragma unroll
    for (int nt = 0; nt < 4; ++nt) {
      const size_t nn = n0 + nt * 16 + (lane & 15);
      if (mt < 8) {
        f32x4 z0, z1;
#pragma unroll
        for (int r = 0; r < 4; ++r) z0[r] = sigmoidf_(acc[i][nt][r]     + bz[m0 + r]);
#pragma unroll
        for (int r = 0; r < 4; ++r) z1[r] = sigmoidf_(acc[i][nt][4 + r] + bz[m0 + 4 + r]);
        float* zp = zbuf + nn * CH + m0;
        *(f32x4*)zp = z0;
        *(f32x4*)(zp + 4) = z1;
      } else {
        const int mm0 = m0 - CH;
        const float* hp = hF + nn * CH + mm0;
        f32x4 h0 = *(const f32x4*)hp;
        f32x4 h1 = *(const f32x4*)(hp + 4);
        int b2 = (int)(nn / HW), r2 = (int)nn - b2 * HW;
        int y2 = r2 / WW, x2 = r2 - y2 * WW;
        size_t p2 = ((size_t)(b2 * HP + y2 + 2)) * WP + (x2 + 2);
        u16x8 o;
#pragma unroll
        for (int r = 0; r < 4; ++r)
          o[r]     = f2bf(sigmoidf_(acc[i][nt][r]     + br[mm0 + r])     * h0[r]);
#pragma unroll
        for (int r = 0; r < 4; ++r)
          o[4 + r] = f2bf(sigmoidf_(acc[i][nt][4 + r] + br[mm0 + 4 + r]) * h1[r]);
        *(u16x8*)&rh[p2 * CH + mm0] = o;
      }
    }
  }
}

// ---------------------------------------------------------------------------
// q conv GEMM + GRU update: M=128, N=128, K=2240.
// reads rh (ch<128, padded NHWC bf16) + act x-channels; writes hF (f32),
// act h-channels (bf16), optionally NCHW d_out.
// ---------------------------------------------------------------------------
template <int VERT>
__global__ __launch_bounds__(256)
void gemm_q(const uint16_t* __restrict__ rh, const uint16_t* __restrict__ act,
            const uint16_t* __restrict__ aFrag, const float* __restrict__ bq,
            const float* __restrict__ zbuf, float* __restrict__ hF,
            uint16_t* __restrict__ actH, float* __restrict__ outNCHW) {
  __shared__ alignas(32) uint16_t lds[2][8 * NTSTRIDE];
  const int tid  = threadIdx.x;
  const int lane = tid & 31;
  const int wave = tid >> 5;
  const int n0   = blockIdx.x * 128;

  // two staging chunks per thread: (kgrpA, nl) and (kgrpA+2, nl)
  const int kgA = tid >> 7;                  // 0..1
  const int kgB = kgA + 2;
  const int nl  = tid & 127;
  const int n  = n0 + nl;
  const int b  = n / HW;
  const int rm = n - b * HW;
  const int y  = rm / WW;
  const int xx = rm - y * WW;
  const size_t ps = ((size_t)(b * HP + y + 2)) * WP + (xx + 2);
  const uint16_t* gLowA  = rh  + ps * CH  + kgA * 8;
  const uint16_t* gHighA = act + ps * CIN + kgA * 8;
  const uint16_t* gLowB  = rh  + ps * CH  + kgB * 8;
  const uint16_t* gHighB = act + ps * CIN + kgB * 8;
  const int fA = bfrag_idx(nl, kgA);
  const int fB = bfrag_idx(nl, kgB);
  uint32_t lA[2], lB[2];
  lA[0] = (uint32_t)(uintptr_t)&lds[0][fA];
  lA[1] = (uint32_t)(uintptr_t)&lds[1][fA];
  lB[0] = (uint32_t)(uintptr_t)&lds[0][fB];
  lB[1] = (uint32_t)(uintptr_t)&lds[1][fB];

  v8f acc[8];
#pragma unroll
  for (int j = 0; j < 8; ++j) acc[j] = (v8f){0,0,0,0,0,0,0,0};

  const v16bf* ag = reinterpret_cast<const v16bf*>(aFrag);

  auto stage = [&](int kb, int buf) {
    int t  = kb / KBPT;
    int c0 = (kb - t * KBPT) * 32;
    int sh = (t - 2) * (VERT ? WP : 1);
    int offLow  = c0 + sh * CH;
    int offHigh = c0 + sh * CIN;
    const uint16_t* gA = (c0 + kgA * 8 < CH) ? (gLowA + offLow) : (gHighA + offHigh);
    const uint16_t* gB = (c0 + kgB * 8 < CH) ? (gLowB + offLow) : (gHighB + offHigh);
    async_cp16(gA, lA[buf]);
    async_cp16(gB, lB[buf]);
  };

  stage(0, 0);

  for (int kb = 0; kb < NKB; ++kb) {
    const int cur = kb & 1;
    if (kb + 1 < NKB) {
      stage(kb + 1, cur ^ 1);
      wait_async_le(2);
    } else {
      wait_async_le(0);
    }
    __syncthreads();

    const uint16_t* lcur = lds[cur];
    v16bf a = ag[((size_t)kb * 8 + wave) * 32 + lane];
#pragma unroll
    for (int nt = 0; nt < 8; ++nt) {
      v16bf bf = *reinterpret_cast<const v16bf*>(&lcur[nt * NTSTRIDE + lane * 16]);
      acc[nt] = __builtin_amdgcn_wmma_f32_16x16x32_bf16(
          false, a, false, bf, (short)0, acc[nt], false, false);
    }
  }

  // epilogue: fused GRU update
  const int m0 = wave * 16 + ((lane >= 16) ? 8 : 0);
  f32x4 bq0 = *(const f32x4*)&bq[m0];
  f32x4 bq1 = *(const f32x4*)&bq[m0 + 4];
#pragma unroll
  for (int nt = 0; nt < 8; ++nt) {
    const size_t nn = n0 + nt * 16 + (lane & 15);
    const float* zp = zbuf + nn * CH + m0;
    float*       hp = hF   + nn * CH + m0;
    f32x4 z0 = *(const f32x4*)zp,       z1 = *(const f32x4*)(zp + 4);
    f32x4 h0 = *(const f32x4*)hp,       h1 = *(const f32x4*)(hp + 4);
    f32x4 o0, o1;
    u16x8 ob;
#pragma unroll
    for (int r = 0; r < 4; ++r) {
      float q = tanhf(acc[nt][r] + bq0[r]);
      o0[r] = (1.0f - z0[r]) * h0[r] + z0[r] * q;
      ob[r] = f2bf(o0[r]);
    }
#pragma unroll
    for (int r = 0; r < 4; ++r) {
      float q = tanhf(acc[nt][4 + r] + bq1[r]);
      o1[r] = (1.0f - z1[r]) * h1[r] + z1[r] * q;
      ob[4 + r] = f2bf(o1[r]);
    }
    *(f32x4*)hp = o0;
    *(f32x4*)(hp + 4) = o1;
    int b2 = (int)(nn / HW), r2 = (int)nn - b2 * HW;
    int y2 = r2 / WW, x2 = r2 - y2 * WW;
    size_t p2 = ((size_t)(b2 * HP + y2 + 2)) * WP + (x2 + 2);
    *(u16x8*)&actH[p2 * CIN + m0] = ob;
    if (outNCHW) {
      size_t base = ((size_t)b2 * CH + m0) * HW + (size_t)y2 * WW + x2;
#pragma unroll
      for (int r = 0; r < 4; ++r) outNCHW[base + (size_t)r * HW] = o0[r];
#pragma unroll
      for (int r = 0; r < 4; ++r) outNCHW[base + (size_t)(4 + r) * HW] = o1[r];
    }
  }
}

// ---------------------------------------------------------------------------
extern "C" void kernel_launch(void* const* d_in, const int* in_sizes, int n_in,
                              void* d_out, int out_size, void* d_ws, size_t ws_size,
                              hipStream_t stream) {
  const float* h   = (const float*)d_in[0];
  const float* x   = (const float*)d_in[1];
  const float* wz1 = (const float*)d_in[2];  const float* bz1 = (const float*)d_in[3];
  const float* wr1 = (const float*)d_in[4];  const float* br1 = (const float*)d_in[5];
  const float* wq1 = (const float*)d_in[6];  const float* bq1 = (const float*)d_in[7];
  const float* wz2 = (const float*)d_in[8];  const float* bz2 = (const float*)d_in[9];
  const float* wr2 = (const float*)d_in[10]; const float* br2 = (const float*)d_in[11];
  const float* wq2 = (const float*)d_in[12]; const float* bq2 = (const float*)d_in[13];
  float* out = (float*)d_out;

  uint8_t* ws = (uint8_t*)d_ws;
  size_t off = 0;
  auto walloc = [&](size_t bytes) -> void* {
    void* p = ws + off;
    off += (bytes + 255) & ~(size_t)255;
    return p;
  };
  const size_t NZR = (size_t)NKB * 16 * 32 * 16 * 2;
  const size_t NQ  = (size_t)NKB * 8  * 32 * 16 * 2;
  const size_t ACT_BYTES = (size_t)BB * HP * WP * CIN * 2;   // ~160 MB
  const size_t RH_BYTES  = (size_t)BB * HP * WP * CH  * 2;   // ~46 MB
  uint16_t* azr1 = (uint16_t*)walloc(NZR);
  uint16_t* aq1  = (uint16_t*)walloc(NQ);
  uint16_t* azr2 = (uint16_t*)walloc(NZR);
  uint16_t* aq2  = (uint16_t*)walloc(NQ);
  uint16_t* act  = (uint16_t*)walloc(ACT_BYTES);
  uint16_t* rh   = (uint16_t*)walloc(RH_BYTES);
  float*    hF   = (float*)   walloc((size_t)NSP * CH * 4);  // 84 MB
  float*    zbuf = (float*)   walloc((size_t)NSP * CH * 4);  // 84 MB

  // zero padded buffers (borders must read as 0)
  hipMemsetAsync(act, 0, ACT_BYTES, stream);
  hipMemsetAsync(rh,  0, RH_BYTES,  stream);

  // weight repack
  int tzr = NKB * 16 * 32, tq = NKB * 8 * 32;
  pack_wfrag<<<(tzr + 255) / 256, 256, 0, stream>>>(wz1, wr1, azr1, 16);
  pack_wfrag<<<(tzr + 255) / 256, 256, 0, stream>>>(wz2, wr2, azr2, 16);
  pack_wfrag<<<(tq  + 255) / 256, 256, 0, stream>>>(wq1, wq1, aq1, 8);
  pack_wfrag<<<(tq  + 255) / 256, 256, 0, stream>>>(wq2, wq2, aq2, 8);

  // activation packs
  pack_act<<<dim3(NSP / 256, 56), 256, 0, stream>>>(h, x, act);
  pack_hF <<<dim3(NSP / 256, 16), 256, 0, stream>>>(h, hF);

  // ---- pass 1: horizontal (1x5) ----
  gemm_zr<0><<<NSP / 64,  256, 0, stream>>>(act, azr1, bz1, br1, hF, zbuf, rh);
  gemm_q <0><<<NSP / 128, 256, 0, stream>>>(rh, act, aq1, bq1, zbuf, hF, act, nullptr);

  // ---- pass 2: vertical (5x1) ----
  gemm_zr<1><<<NSP / 64,  256, 0, stream>>>(act, azr2, bz2, br2, hF, zbuf, rh);
  gemm_q <1><<<NSP / 128, 256, 0, stream>>>(rh, act, aq2, bq2, zbuf, hF, act, out);
}
